// TargetCentricAttention_18691697672866
// MI455X (gfx1250) — matrix operand order repaired and checked by reading, not compiled, not run on hardware
//
#include <hip/hip_runtime.h>
#include <hip/hip_bf16.h>

// ---------------------------------------------------------------------------
// Problem constants (reference: B=4, S=2048, D=1024, fp32)
// ---------------------------------------------------------------------------
#define BB 4
#define SS 2048
#define DD 1024
#define BS (BB * SS) // 8192 tokens

typedef __attribute__((ext_vector_type(16))) __bf16 bf16x16;
typedef __attribute__((ext_vector_type(8)))  float  f32x8;
typedef __attribute__((ext_vector_type(4)))  unsigned int u32x4;
typedef int i32x4v __attribute__((vector_size(16)));   // matches async-LDS builtin param

// gfx1250 async global->LDS copy path (ASYNCcnt-tracked), guarded so the
// file still compiles if the toolchain lacks the builtins.
#if defined(__has_builtin)
#if __has_builtin(__builtin_amdgcn_global_load_async_to_lds_b128) && \
    __has_builtin(__builtin_amdgcn_s_wait_asynccnt)
#define HAVE_ASYNC_LDS 1
#endif
#endif

// Round-to-nearest-even f32 -> bf16 (bit pattern)
__device__ __forceinline__ unsigned short f2bf(float f) {
    unsigned u = __float_as_uint(f);
    u += 0x7FFFu + ((u >> 16) & 1u);
    return (unsigned short)(u >> 16);
}

// ---------------------------------------------------------------------------
// WMMA fragment loads, matching CDNA5 ISA VGPR layouts (CDNA5 ISA)
//
// A (16x32 bf16): lane l: m = l&15, half h = l>>4.
//   VGPRs 0..3 hold K = 8h .. 8h+7 (contiguous), VGPRs 4..7 hold K+16.
//   => two contiguous 16-byte loads per lane.
// B (32x16 bf16): lane l: n = l&15, half h = l>>4; same K order as A
//   (elements 0..7: K = 8h..8h+7, elements 8..15: K+16) => strided scalar loads.
// C/D (16x16 f32): lane l: n = l&15; VGPR i: m = i + 8*(l>>4).
// ---------------------------------------------------------------------------
__device__ __forceinline__ bf16x16 ldA(const unsigned short* __restrict__ base,
                                       int lda, int kb) {
    unsigned l = threadIdx.x & 31u;
    unsigned h = l >> 4;
    unsigned m = l & 15u;
    const unsigned short* p = base + (size_t)m * (size_t)lda + (size_t)(kb + (int)(h * 8u));
    union { u32x4 u[2]; bf16x16 v; } t;
    t.u[0] = *(const u32x4*)(p);
    t.u[1] = *(const u32x4*)(p + 16);
    return t.v;
}

__device__ __forceinline__ bf16x16 ldB(const unsigned short* __restrict__ base,
                                       int ldb, int kb) {
    unsigned l = threadIdx.x & 31u;
    unsigned h = l >> 4;
    unsigned n = l & 15u;
    const unsigned short* p = base + (size_t)(kb + (int)(h * 8u)) * (size_t)ldb + n;
    union { unsigned short s[16]; bf16x16 v; } t;
#pragma unroll
    for (int s = 0; s < 8; s++) {
        t.s[s] = p[(size_t)s * (size_t)ldb];
        t.s[8 + s] = p[(size_t)(16 + s) * (size_t)ldb];
    }
    return t.v;
}

#define WMMA_BF16(A, B, C) \
    __builtin_amdgcn_wmma_f32_16x16x32_bf16(false, (A), false, (B), (short)0, (C), false, false)

// ---------------------------------------------------------------------------
// K0a: convert x (f32) -> bf16, elementwise
// ---------------------------------------------------------------------------
__global__ __launch_bounds__(256) void k_conv_x(const float* __restrict__ x,
                                                unsigned short* __restrict__ xbf) {
    size_t n = (size_t)BS * DD;
    for (size_t i = (size_t)blockIdx.x * blockDim.x + threadIdx.x; i < n;
         i += (size_t)gridDim.x * blockDim.x) {
        xbf[i] = f2bf(x[i]);
    }
}

// ---------------------------------------------------------------------------
// K0b: Wt[z][d][e] = bf16(W_z[e][d])  (transpose so GEMM B-operand is K-major)
// ---------------------------------------------------------------------------
__global__ __launch_bounds__(256) void k_conv_w(const float* __restrict__ Wq,
                                                const float* __restrict__ Wk,
                                                const float* __restrict__ Wv,
                                                unsigned short* __restrict__ wt) {
    int z = blockIdx.z;
    const float* W = (z == 0) ? Wq : (z == 1) ? Wk : Wv;
    unsigned short* dst = wt + (size_t)z * DD * DD;
    size_t n = (size_t)DD * DD;
    for (size_t i = (size_t)blockIdx.x * blockDim.x + threadIdx.x; i < n;
         i += (size_t)gridDim.x * blockDim.x) {
        int d = (int)(i / DD);
        int e = (int)(i % DD);
        dst[i] = f2bf(W[(size_t)e * DD + d]);
    }
}

// ---------------------------------------------------------------------------
// K1: QKV projection.  y = x @ W^T + b  for z in {q,k,v}.
//   A = xbf [BS x D], B = Wt_z [D x D] (K-major).  128x128 block / 64x64 wave.
//   z==0 (q): stored row-major, pre-scaled by 1/sqrt(D)=1/32 (folds into scores)
//   z==1 (k): stored TRANSPOSED per batch: kT[b][d][s]  (16B packed store/lane)
//   z==2 (v): stored row-major
// ---------------------------------------------------------------------------
__global__ __launch_bounds__(128) void k_qkv(const unsigned short* __restrict__ xbf,
                                             const unsigned short* __restrict__ wt,
                                             const float* __restrict__ bq,
                                             const float* __restrict__ bk,
                                             const float* __restrict__ bv,
                                             unsigned short* __restrict__ qbf,
                                             unsigned short* __restrict__ kT,
                                             unsigned short* __restrict__ vbf) {
    const int z = blockIdx.z;
    const int w = threadIdx.x >> 5;
    const int Mb = blockIdx.x * 128 + (w >> 1) * 64;
    const int Nb = blockIdx.y * 128 + (w & 1) * 64;
    const unsigned short* Abase = xbf + (size_t)Mb * DD;
    const unsigned short* Bbase = wt + (size_t)z * DD * DD + Nb;
    const float* bias = (z == 0) ? bq : (z == 1) ? bk : bv;
    const unsigned lpf = threadIdx.x & 31u;

    f32x8 c[4][4];
#pragma unroll
    for (int i = 0; i < 4; i++)
#pragma unroll
        for (int j = 0; j < 4; j++) c[i][j] = (f32x8)0.0f;

#pragma unroll 1
    for (int kb = 0; kb < DD; kb += 32) {
        // near-scope speculative prefetch of the operand stream, 2 K-steps ahead
        __builtin_prefetch(Abase + (size_t)lpf * DD + kb + 64, 0, 3);
        __builtin_prefetch(Bbase + (size_t)(kb + 64 + (int)lpf) * DD, 0, 3);
        bf16x16 a[4];
#pragma unroll
        for (int i = 0; i < 4; i++) a[i] = ldA(Abase + (size_t)(i * 16) * DD, DD, kb);
#pragma unroll
        for (int j = 0; j < 4; j++) {
            bf16x16 b = ldB(Bbase + j * 16, DD, kb);
#pragma unroll
            for (int i = 0; i < 4; i++) c[i][j] = WMMA_BF16(a[i], b, c[i][j]);
        }
    }

    const unsigned l = threadIdx.x & 31u;
    const unsigned h = l >> 4;
    const unsigned nn = l & 15u;
    const float scale = (z == 0) ? 0.03125f : 1.0f;

#pragma unroll
    for (int j = 0; j < 4; j++) {
        int n = Nb + j * 16 + (int)nn;
        float bs = bias[n];
#pragma unroll
        for (int i = 0; i < 4; i++) {
            int m0 = Mb + i * 16 + (int)(h * 8u);
            if (z == 1) {
                // transposed store: lane owns 8 consecutive s positions of row d=n
                u32x4 pk;
#pragma unroll
                for (int p2 = 0; p2 < 4; p2++) {
                    float e0 = c[i][j][2 * p2] + bs;
                    float e1 = c[i][j][2 * p2 + 1] + bs;
                    pk[p2] = (unsigned)f2bf(e0) | ((unsigned)f2bf(e1) << 16);
                }
                size_t bidx = (size_t)(m0 >> 11);           // batch = token/2048
                size_t s0 = (size_t)(m0 & 2047);
                *(u32x4*)(kT + bidx * (size_t)DD * SS + (size_t)n * SS + s0) = pk;
            } else {
                unsigned short* dst = (z == 0) ? qbf : vbf;
#pragma unroll
                for (int ii = 0; ii < 8; ii++)
                    dst[(size_t)(m0 + ii) * DD + n] = f2bf((c[i][j][ii] + bs) * scale);
            }
        }
    }
}

// ---------------------------------------------------------------------------
// K2: scores[b][q][k] = sum_d qbf[b][q][d] * kT[b][d][k]   (q already /32)
// ---------------------------------------------------------------------------
__global__ __launch_bounds__(128) void k_scores(const unsigned short* __restrict__ qbf,
                                                const unsigned short* __restrict__ kT,
                                                float* __restrict__ scores) {
    const int b = blockIdx.z;
    const int w = threadIdx.x >> 5;
    const int Mb = blockIdx.x * 128 + (w >> 1) * 64;
    const int Nb = blockIdx.y * 128 + (w & 1) * 64;
    const unsigned short* Abase = qbf + (size_t)b * SS * DD + (size_t)Mb * DD;
    const unsigned short* Bbase = kT + (size_t)b * DD * SS + Nb;
    float* out = scores + (size_t)b * SS * SS;
    const unsigned lpf = threadIdx.x & 31u;

    f32x8 c[4][4];
#pragma unroll
    for (int i = 0; i < 4; i++)
#pragma unroll
        for (int j = 0; j < 4; j++) c[i][j] = (f32x8)0.0f;

#pragma unroll 1
    for (int kb = 0; kb < DD; kb += 32) {
        __builtin_prefetch(Abase + (size_t)lpf * DD + kb + 64, 0, 3);
        __builtin_prefetch(Bbase + (size_t)(kb + 64 + (int)lpf) * SS, 0, 3);
        bf16x16 a[4];
#pragma unroll
        for (int i = 0; i < 4; i++) a[i] = ldA(Abase + (size_t)(i * 16) * DD, DD, kb);
#pragma unroll
        for (int j = 0; j < 4; j++) {
            bf16x16 bfr = ldB(Bbase + j * 16, SS, kb);
#pragma unroll
            for (int i = 0; i < 4; i++) c[i][j] = WMMA_BF16(a[i], bfr, c[i][j]);
        }
    }

    const unsigned l = threadIdx.x & 31u;
    const unsigned h = l >> 4;
    const unsigned nn = l & 15u;
#pragma unroll
    for (int j = 0; j < 4; j++) {
        int n = Nb + j * 16 + (int)nn;
#pragma unroll
        for (int i = 0; i < 4; i++) {
            int m0 = Mb + i * 16 + (int)(h * 8u);
#pragma unroll
            for (int ii = 0; ii < 8; ii++)
                out[(size_t)(m0 + ii) * SS + n] = c[i][j][ii];
        }
    }
}

// ---------------------------------------------------------------------------
// K3: row softmax over k; write probsT[b][k][q] (bf16) so K4's A-operand is
//     a plain row-major read.  The 8KB score row is staged into LDS once via
//     CDNA5 async global->LDS copies (ASYNCcnt) and consumed by both passes.
// ---------------------------------------------------------------------------
__global__ __launch_bounds__(256) void k_softmax(const float* __restrict__ scores,
                                                 unsigned short* __restrict__ probsT) {
    const int q = blockIdx.x;
    const int b = blockIdx.y;
    const int tid = threadIdx.x;
    const float* row = scores + ((size_t)b * SS + q) * SS;

    __shared__ float rowbuf[SS];
    __shared__ float red[256];

#ifdef HAVE_ASYNC_LDS
    // 2048 f32 = 8KB: each of 256 threads async-copies 2 x 16B, LDS-side
    // addresses mirror global so the row lands contiguously.
    __builtin_amdgcn_global_load_async_to_lds_b128(
        (i32x4v*)(row + tid * 4), (i32x4v*)(&rowbuf[tid * 4]), 0, 0);
    __builtin_amdgcn_global_load_async_to_lds_b128(
        (i32x4v*)(row + 1024 + tid * 4), (i32x4v*)(&rowbuf[1024 + tid * 4]), 0, 0);
    __builtin_amdgcn_s_wait_asynccnt(0);
    __syncthreads();
#else
#pragma unroll
    for (int e = 0; e < 8; e++) rowbuf[tid + e * 256] = row[tid + e * 256];
    __syncthreads();
#endif

    float v[8];
    float mx = -3.4e38f;
#pragma unroll
    for (int e = 0; e < 8; e++) {
        v[e] = rowbuf[tid + e * 256];
        mx = fmaxf(mx, v[e]);
    }

    red[tid] = mx;
    __syncthreads();
    for (int s = 128; s > 0; s >>= 1) {
        if (tid < s) red[tid] = fmaxf(red[tid], red[tid + s]);
        __syncthreads();
    }
    mx = red[0];
    __syncthreads();

    float sum = 0.0f;
#pragma unroll
    for (int e = 0; e < 8; e++) {
        v[e] = __expf(v[e] - mx);
        sum += v[e];
    }
    red[tid] = sum;
    __syncthreads();
    for (int s = 128; s > 0; s >>= 1) {
        if (tid < s) red[tid] += red[tid + s];
        __syncthreads();
    }
    float inv = 1.0f / red[0];

#pragma unroll
    for (int e = 0; e < 8; e++) {
        int k = tid + e * 256;
        probsT[((size_t)b * SS + k) * SS + q] = f2bf(v[e] * inv);
    }
}

// ---------------------------------------------------------------------------
// K4: out[b][k][d] = sum_q probsT[b][k][q] * vbf[b][q][d]
// ---------------------------------------------------------------------------
__global__ __launch_bounds__(128) void k_out(const unsigned short* __restrict__ probsT,
                                             const unsigned short* __restrict__ vbf,
                                             float* __restrict__ out) {
    const int b = blockIdx.z;
    const int w = threadIdx.x >> 5;
    const int Mb = blockIdx.x * 128 + (w >> 1) * 64;
    const int Nb = blockIdx.y * 128 + (w & 1) * 64;
    const unsigned short* Abase = probsT + (size_t)b * SS * SS + (size_t)Mb * SS;
    const unsigned short* Bbase = vbf + (size_t)b * SS * DD + Nb;
    float* obase = out + (size_t)b * SS * DD;
    const unsigned lpf = threadIdx.x & 31u;

    f32x8 c[4][4];
#pragma unroll
    for (int i = 0; i < 4; i++)
#pragma unroll
        for (int j = 0; j < 4; j++) c[i][j] = (f32x8)0.0f;

#pragma unroll 1
    for (int kb = 0; kb < SS; kb += 32) {
        __builtin_prefetch(Abase + (size_t)lpf * SS + kb + 64, 0, 3);
        __builtin_prefetch(Bbase + (size_t)(kb + 64 + (int)lpf) * DD, 0, 3);
        bf16x16 a[4];
#pragma unroll
        for (int i = 0; i < 4; i++) a[i] = ldA(Abase + (size_t)(i * 16) * SS, SS, kb);
#pragma unroll
        for (int j = 0; j < 4; j++) {
            bf16x16 bfr = ldB(Bbase + j * 16, DD, kb);
#pragma unroll
            for (int i = 0; i < 4; i++) c[i][j] = WMMA_BF16(a[i], bfr, c[i][j]);
        }
    }

    const unsigned l = threadIdx.x & 31u;
    const unsigned h = l >> 4;
    const unsigned nn = l & 15u;
#pragma unroll
    for (int j = 0; j < 4; j++) {
        int n = Nb + j * 16 + (int)nn;
#pragma unroll
        for (int i = 0; i < 4; i++) {
            int m0 = Mb + i * 16 + (int)(h * 8u);
#pragma unroll
            for (int ii = 0; ii < 8; ii++)
                obase[(size_t)(m0 + ii) * DD + n] = c[i][j][ii];
        }
    }
}

// ---------------------------------------------------------------------------
// Launch: workspace layout (bytes)
//   xbf    16 MB | wt 6 MB | qbf 16 MB | kT 16 MB | vbf 16 MB
//   scores 64 MB | probsT 32 MB                      (~166 MB total, L2 = 192MB)
// ---------------------------------------------------------------------------
extern "C" void kernel_launch(void* const* d_in, const int* in_sizes, int n_in,
                              void* d_out, int out_size, void* d_ws, size_t ws_size,
                              hipStream_t stream) {
    (void)in_sizes; (void)n_in; (void)out_size; (void)ws_size;

    const float* x  = (const float*)d_in[0];
    const float* Wq = (const float*)d_in[1];
    const float* bq = (const float*)d_in[2];
    const float* Wk = (const float*)d_in[3];
    const float* bk = (const float*)d_in[4];
    const float* Wv = (const float*)d_in[5];
    const float* bv = (const float*)d_in[6];
    float* out = (float*)d_out;

    char* ws = (char*)d_ws;
    const size_t sz_xbf = (size_t)BS * DD * 2;       // 16 MB
    const size_t sz_wt  = (size_t)3 * DD * DD * 2;   //  6 MB
    const size_t sz_q   = (size_t)BS * DD * 2;       // 16 MB
    const size_t sz_kT  = (size_t)BB * DD * SS * 2;  // 16 MB
    const size_t sz_v   = (size_t)BS * DD * 2;       // 16 MB
    const size_t sz_sc  = (size_t)BB * SS * SS * 4;  // 64 MB

    unsigned short* xbf    = (unsigned short*)(ws);
    unsigned short* wt     = (unsigned short*)(ws + sz_xbf);
    unsigned short* qbf    = (unsigned short*)(ws + sz_xbf + sz_wt);
    unsigned short* kT     = (unsigned short*)(ws + sz_xbf + sz_wt + sz_q);
    unsigned short* vbf    = (unsigned short*)(ws + sz_xbf + sz_wt + sz_q + sz_kT);
    float*          scores = (float*)        (ws + sz_xbf + sz_wt + sz_q + sz_kT + sz_v);
    unsigned short* probsT = (unsigned short*)(ws + sz_xbf + sz_wt + sz_q + sz_kT + sz_v + sz_sc);

    // K0: precision conversion + weight transpose
    k_conv_x<<<dim3(4096), dim3(256), 0, stream>>>(x, xbf);
    k_conv_w<<<dim3(1024, 1, 3), dim3(256), 0, stream>>>(Wq, Wk, Wv, wt);

    // K1: QKV projection (8192x1024, K=1024) x3
    k_qkv<<<dim3(BS / 128, DD / 128, 3), dim3(128), 0, stream>>>(
        xbf, wt, bq, bk, bv, qbf, kT, vbf);

    // K2: scores (2048x2048, K=1024) per batch
    k_scores<<<dim3(SS / 128, SS / 128, BB), dim3(128), 0, stream>>>(qbf, kT, scores);

    // K3: softmax rows -> probsT (bf16)
    k_softmax<<<dim3(SS, BB), dim3(256), 0, stream>>>(scores, probsT);

    // K4: out = probsT @ v (2048x1024, K=2048) per batch
    k_out<<<dim3(SS / 128, DD / 128, BB), dim3(128), 0, stream>>>(probsT, vbf, out);
}
